// MaskVectorQuantize_47768626266372
// MI455X (gfx1250) — matrix operand order repaired and checked by reading, not compiled, not run on hardware
//
#include <hip/hip_runtime.h>
#include <hip/hip_bf16.h>

typedef __attribute__((ext_vector_type(16))) __bf16 v16bf;
typedef __attribute__((ext_vector_type(8)))  float  v8f;

#define N_TOK   16384            // B*H*W tokens
#define K_CB    16384            // codebook size
#define C_DIM   256              // embedding dim
#define XQ_ELEMS (16*256*32*32)  // 4194304 output elements for xq

// workspace layout (bytes)
#define WHI_OFF 0u
#define WLO_OFF (8u*1024u*1024u)
#define HN_OFF  (16u*1024u*1024u)
#define IND_OFF (HN_OFF + K_CB*4u)
#define ACC_OFF (IND_OFF + N_TOK*4u)

// ---------------------------------------------------------------------------
// Phase 1: split weight into bf16 hi/lo, compute 0.5*||e_k||^2, zero accums.
// ---------------------------------------------------------------------------
__global__ __launch_bounds__(256) void vq_prep(const float* __restrict__ weight,
                                               __bf16* __restrict__ whi,
                                               __bf16* __restrict__ wlo,
                                               float* __restrict__ halfnorm,
                                               float* __restrict__ accum) {
    const int k = blockIdx.x, c = threadIdx.x;
    const size_t idx = (size_t)k * C_DIM + c;
    float w = weight[idx];
    __bf16 hi = (__bf16)w;
    whi[idx] = hi;
    wlo[idx] = (__bf16)(w - (float)hi);

    float sq = w * w;
    #pragma unroll
    for (int off = 16; off > 0; off >>= 1) sq += __shfl_xor(sq, off, 32);
    __shared__ float red[8];
    const int wv = threadIdx.x >> 5, ln = threadIdx.x & 31;
    if (ln == 0) red[wv] = sq;
    __syncthreads();
    if (threadIdx.x == 0) {
        float t = 0.f;
        #pragma unroll
        for (int i = 0; i < 8; ++i) t += red[i];
        halfnorm[k] = 0.5f * t;
    }
    if (blockIdx.x == 0 && threadIdx.x < 2) accum[threadIdx.x] = 0.f;
}

// ---------------------------------------------------------------------------
// Phase 2: fused GEMM + argmax. One wave32 owns 16 token rows; A fragments
// (bf16 hi/lo, 8 k-steps) live in VGPRs for the whole codebook scan. Each
// 16-column codebook tile costs 24 v_wmma_f32_16x16x32_bf16 (3-term split).
// ---------------------------------------------------------------------------
__global__ __launch_bounds__(256) void vq_argmax(const float* __restrict__ x,
                                                 const __bf16* __restrict__ whi,
                                                 const __bf16* __restrict__ wlo,
                                                 const float* __restrict__ halfnorm,
                                                 int* __restrict__ ind) {
    const int wave = threadIdx.x >> 5;
    const int lane = threadIdx.x & 31;
    const int g    = lane >> 4;        // lane half (ISA A/B layout)
    const int nn   = lane & 15;        // row-in-tile (A) / col-in-tile (B,C)
    const int rowbase = blockIdx.x * 128 + wave * 16;

    // Gather this wave's 16 token rows from x (B,C,H,W) with the NHWC
    // transpose folded into the addressing, split into bf16 hi/lo fragments
    // in contiguous k order, k = 16*g + j within each 32-wide k-step (16x32 per k-step).
    const int n  = rowbase + nn;
    const int b  = n >> 10, hw = n & 1023;
    const float* xb = x + (size_t)b * (C_DIM * 1024) + hw;

    v16bf a_hi[8], a_lo[8];
    #pragma unroll
    for (int s = 0; s < 8; ++s) {
        #pragma unroll
        for (int j = 0; j < 16; ++j) {
            const int c = 32 * s + 16 * g + j;
            const float v = xb[(size_t)c * 1024];
            const __bf16 h = (__bf16)v;
            a_hi[s][j] = h;
            a_lo[s][j] = (__bf16)(v - (float)h);
        }
    }

    float best[8];
    int   bidx[8];
    #pragma unroll
    for (int r = 0; r < 8; ++r) { best[r] = -3.0e38f; bidx[r] = 0; }

    for (int t = 0; t < K_CB / 16; ++t) {
        const int col = t * 16 + nn;                       // codebook index
        const size_t rowoff = (size_t)col * C_DIM + 16 * g;
        v8f acc = {0.f, 0.f, 0.f, 0.f, 0.f, 0.f, 0.f, 0.f};
        #pragma unroll
        for (int s = 0; s < 8; ++s) {
            const v16bf bh = *(const v16bf*)(whi + rowoff + 32 * s);
            const v16bf bl = *(const v16bf*)(wlo + rowoff + 32 * s);
            acc = __builtin_amdgcn_wmma_f32_16x16x32_bf16(false, a_hi[s], false, bh,
                                                          (short)0, acc, false, false);
            acc = __builtin_amdgcn_wmma_f32_16x16x32_bf16(false, a_hi[s], false, bl,
                                                          (short)0, acc, false, false);
            acc = __builtin_amdgcn_wmma_f32_16x16x32_bf16(false, a_lo[s], false, bh,
                                                          (short)0, acc, false, false);
        }
        const float hn = halfnorm[col];
        #pragma unroll
        for (int r = 0; r < 8; ++r) {
            const float cand = acc[r] - hn;                // x.e - 0.5||e||^2
            if (cand > best[r]) { best[r] = cand; bidx[r] = col; }
        }
        if (t + 1 < K_CB / 16)
            __builtin_prefetch(whi + ((size_t)(t + 1) * 16 + nn) * C_DIM, 0, 1);
    }

    // C/D layout: lane element r holds row rowbase + r + 8*g, column class nn.
    // Butterfly-reduce (max, lowest-index ties) across the 16 column lanes.
    #pragma unroll
    for (int r = 0; r < 8; ++r) {
        float v = best[r];
        int   id = bidx[r];
        #pragma unroll
        for (int off = 1; off < 16; off <<= 1) {
            const float ov = __shfl_xor(v, off, 32);
            const int   oi = __shfl_xor(id, off, 32);
            if (ov > v || (ov == v && oi < id)) { v = ov; id = oi; }
        }
        if (nn == 0) ind[rowbase + 8 * g + r] = id;
    }
}

// ---------------------------------------------------------------------------
// Phase 3: gather weight[ind] -> xq (straight-through value == xq), scatter
// back to (B,C,H,W), accumulate masked squared error S and sum(mask).
// loss = (1 + BETA) * S / (C * sum(mask)) = 1.25 * S / (256 * sum_mask)
// ---------------------------------------------------------------------------
__global__ __launch_bounds__(256) void vq_finalize(const float* __restrict__ x,
                                                   const float* __restrict__ cmask,
                                                   const float* __restrict__ weight,
                                                   const int* __restrict__ ind,
                                                   float* __restrict__ out,
                                                   float* __restrict__ accum) {
    const int n = blockIdx.x, c = threadIdx.x;
    const int k = ind[n];
    const int b = n >> 10, hw = n & 1023;
    const float wv = weight[(size_t)k * C_DIM + c];
    const size_t xi = ((size_t)b * C_DIM + c) * 1024 + hw;
    const float xv = x[xi];
    out[xi] = wv;

    const float m = cmask[n];
    const float d = wv - xv;
    float val = m * d * d;
    #pragma unroll
    for (int off = 16; off > 0; off >>= 1) val += __shfl_xor(val, off, 32);
    __shared__ float red[8];
    const int wvi = threadIdx.x >> 5, ln = threadIdx.x & 31;
    if (ln == 0) red[wvi] = val;
    __syncthreads();
    if (threadIdx.x == 0) {
        float t = 0.f;
        #pragma unroll
        for (int i = 0; i < 8; ++i) t += red[i];
        atomicAdd(&accum[0], t);
        atomicAdd(&accum[1], m);
        out[XQ_ELEMS + 1 + n] = (float)k;   // ind_out as float
    }
}

__global__ void vq_loss(const float* __restrict__ accum, float* __restrict__ out) {
    out[XQ_ELEMS] = 1.25f * accum[0] / (256.0f * accum[1]);
}

// ---------------------------------------------------------------------------
extern "C" void kernel_launch(void* const* d_in, const int* in_sizes, int n_in,
                              void* d_out, int out_size, void* d_ws, size_t ws_size,
                              hipStream_t stream) {
    (void)in_sizes; (void)n_in; (void)out_size; (void)ws_size;
    const float* x      = (const float*)d_in[0];   // (16,256,32,32)
    const float* cmask  = (const float*)d_in[1];   // (16,1,32,32)
    const float* weight = (const float*)d_in[2];   // (16384,256)
    float* out = (float*)d_out;

    char* ws = (char*)d_ws;
    __bf16* whi      = (__bf16*)(ws + WHI_OFF);
    __bf16* wlo      = (__bf16*)(ws + WLO_OFF);
    float*  halfnorm = (float*)(ws + HN_OFF);
    int*    ind      = (int*)(ws + IND_OFF);
    float*  accum    = (float*)(ws + ACC_OFF);

    vq_prep   <<<K_CB,        256, 0, stream>>>(weight, whi, wlo, halfnorm, accum);
    vq_argmax <<<N_TOK / 128, 256, 0, stream>>>(x, whi, wlo, halfnorm, ind);
    vq_finalize<<<N_TOK,      256, 0, stream>>>(x, cmask, weight, ind, out, accum);
    vq_loss   <<<1,           1,   0, stream>>>(accum, out);
}
